// Recurrence_44289702756863
// MI455X (gfx1250) — matrix operand order repaired and checked by reading, not compiled, not run on hardware
//
#include <hip/hip_runtime.h>

#define T_DIM 128
#define N_DIM 1024
#define H_DIM 256
#define L_DIM 64
#define C_DIM 64
#define OUT_STRIDE 386   // 1 + 1 + 256 + 64 + 64

typedef __bf16 bf16_t;
typedef __attribute__((ext_vector_type(16))) __bf16 v16bf;
typedef __attribute__((ext_vector_type(8)))  __bf16 v8bf;
typedef __attribute__((ext_vector_type(8)))  float  v8f;

// ---------------------------------------------------------------------------
// f32 -> bf16 converter (weights)
// ---------------------------------------------------------------------------
__global__ void k_f32_to_bf16(const float* __restrict__ in, bf16_t* __restrict__ out, int n) {
    int i = blockIdx.x * blockDim.x + threadIdx.x;
    if (i < n) out[i] = (bf16_t)in[i];
}

// ---------------------------------------------------------------------------
// M = emb[lines]  -> bf16 [N, L, H]
// ---------------------------------------------------------------------------
__global__ void k_embed(const float* __restrict__ emb, const int* __restrict__ lines,
                        bf16_t* __restrict__ Mbf) {
    int i = blockIdx.x * blockDim.x + threadIdx.x;  // over N*L*H
    int h  = i & (H_DIM - 1);
    int nl = i >> 8;                                // H == 256
    int tok = lines[nl];
    Mbf[i] = (bf16_t)emb[(long)tok * H_DIM + h];
}

// ---------------------------------------------------------------------------
// WMMA GEMM:  C[M,Nc] = A[M,K](bf16, row stride lda) * W[Nc,K]^T (bf16) + bias
// One wave computes a 16x32 super-tile (two 16x16 WMMA tiles sharing the A
// fragment). 8 waves/block -> 64x64 block tile. Double-buffered fragments so
// the scheduler can overlap L2 fetch of k+32 with the XDL op on k.
// Fragment layouts used here (wave32; A/B use contiguous k, not the ISA table order):
//   A 16x32 bf16 : lanes 0-15 row=lane, K=0..15; lanes 16-31 K=16..31
//   B 32x16 bf16 : lanes 0-15 col=lane, K=0..15; lanes 16-31 K=16..31
//   C 16x16 f32  : vgpr j -> row j + 8*(lane>=16), col = lane&15
// ---------------------------------------------------------------------------
__global__ __launch_bounds__(256)
void k_wmma_gemm(const bf16_t* __restrict__ A, long lda,
                 const bf16_t* __restrict__ W, const float* __restrict__ bias,
                 int M, int Nc, int K,
                 float* __restrict__ outF, bf16_t* __restrict__ outB,
                 int ldo, int relu)
{
    const int wave = threadIdx.x >> 5;
    const int lane = threadIdx.x & 31;
    const int m0 = (blockIdx.x * 4 + (wave & 3)) * 16;
    const int n0 = (blockIdx.y * 2 + (wave >> 2)) * 32;   // 32-wide super-tile
    if (m0 >= M || n0 >= Nc) return;                      // wave-uniform branch
    const int half = lane >> 4;
    const int l15  = lane & 15;

    const bf16_t* arow  = A + (long)(m0 + l15) * lda;
    const bf16_t* wrow0 = W + (long)(n0 + l15) * K + half * 16;
    const bf16_t* wrow1 = wrow0 + (long)16 * K;

    v8f acc0 = {}, acc1 = {};

    auto loadA = [&](int k0, v16bf& a) {
#pragma unroll
        for (int i = 0; i < 16; ++i) {
            a[i] = arow[k0 + half * 16 + i];
        }
    };
    auto loadB = [&](const bf16_t* wr, int k0, v16bf& b) {
#pragma unroll
        for (int i = 0; i < 16; ++i) b[i] = wr[k0 + i];
    };

    v16bf a0, b00, b10;
    loadA(0, a0);
    loadB(wrow0, 0, b00);
    loadB(wrow1, 0, b10);

    int k0 = 0;
    for (; k0 + 32 < K; k0 += 32) {
        __builtin_prefetch(arow + k0 + 96, 0, 1);
        v16bf a1, b01, b11;
        loadA(k0 + 32, a1);
        loadB(wrow0, k0 + 32, b01);
        loadB(wrow1, k0 + 32, b11);
        acc0 = __builtin_amdgcn_wmma_f32_16x16x32_bf16(false, a0, false, b00, (short)0, acc0, false, false);
        acc1 = __builtin_amdgcn_wmma_f32_16x16x32_bf16(false, a0, false, b10, (short)0, acc1, false, false);
        a0 = a1; b00 = b01; b10 = b11;
    }
    acc0 = __builtin_amdgcn_wmma_f32_16x16x32_bf16(false, a0, false, b00, (short)0, acc0, false, false);
    acc1 = __builtin_amdgcn_wmma_f32_16x16x32_bf16(false, a0, false, b10, (short)0, acc1, false, false);

    const int col0 = n0 + l15;
    const int col1 = col0 + 16;
    const float bv0 = bias ? bias[col0] : 0.0f;
    const float bv1 = bias ? bias[col1] : 0.0f;
#pragma unroll
    for (int j = 0; j < 8; ++j) {
        const int row = m0 + j + half * 8;
        float v0 = acc0[j] + bv0;
        float v1 = acc1[j] + bv1;
        if (relu) { v0 = fmaxf(v0, 0.0f); v1 = fmaxf(v1, 0.0f); }
        if (outF) {
            outF[(long)row * ldo + col0] = v0;
            outF[(long)row * ldo + col1] = v1;
        }
        if (outB) {
            outB[(long)row * ldo + col0] = (bf16_t)v0;
            outB[(long)row * ldo + col1] = (bf16_t)v1;
        }
    }
}

// ---------------------------------------------------------------------------
// GRU gate math (PyTorch order r,z,n). In-place h update; writes bf16 copy,
// optional K slice (BiGRU) and optional h into the packed output (time loop).
// ---------------------------------------------------------------------------
__global__ void k_gru_gate(const float* __restrict__ gi, const float* __restrict__ gh,
                           float* __restrict__ h, bf16_t* __restrict__ hb,
                           bf16_t* __restrict__ Kslice,   // base pre-offset by dir*H, may be null
                           int l,
                           float* __restrict__ outh)      // base = out + t*N*386 + 2, may be null
{
    int i = blockIdx.x * blockDim.x + threadIdx.x;  // over N*H
    int n = i >> 8;
    int j = i & 255;
    long g = (long)n * (3 * H_DIM) + j;
    float ir = gi[g], iz = gi[g + H_DIM], in_ = gi[g + 2 * H_DIM];
    float hr = gh[g], hz = gh[g + H_DIM], hn  = gh[g + 2 * H_DIM];
    float r  = 1.0f / (1.0f + expf(-(ir + hr)));
    float z  = 1.0f / (1.0f + expf(-(iz + hz)));
    float nn = tanhf(in_ + r * hn);
    float hv = (1.0f - z) * nn + z * h[i];
    h[i]  = hv;
    hb[i] = (bf16_t)hv;
    if (Kslice) Kslice[(long)n * (L_DIM * 2 * H_DIM) + (long)l * (2 * H_DIM) + j] = (bf16_t)hv;
    if (outh)   outh[(long)n * OUT_STRIDE + j] = hv;
}

// ---------------------------------------------------------------------------
// x0 = [cond_t, M[n, prev_action]]  (bf16, width C+H = 320)
// ---------------------------------------------------------------------------
__global__ void k_build_x0(const float* __restrict__ cond_t, const int* __restrict__ actions,
                           int t, const bf16_t* __restrict__ Mbf, bf16_t* __restrict__ x0)
{
    int i = blockIdx.x * blockDim.x + threadIdx.x;  // over N*320
    int n = i / (C_DIM + H_DIM);
    int j = i % (C_DIM + H_DIM);
    if (j < C_DIM) {
        x0[i] = (bf16_t)cond_t[(long)n * C_DIM + j];
    } else {
        int a = (t > 0) ? actions[(long)(t - 1) * N_DIM + n] : 0;
        x0[i] = Mbf[((long)n * L_DIM + a) * H_DIM + (j - C_DIM)];
    }
}

// ---------------------------------------------------------------------------
// Head: logits w[n,l] = <K[n,l,:], kvec[n,:]>, softmax over L, critic value,
// pack a/v/probs/p into out[t,n,:]; optionally also write the hx[-1:] tail.
// One block per batch element n; 64 threads (one per line l).
// K reads vectorized as 16-byte v8bf loads (L2-bandwidth-bound inner loop).
// ---------------------------------------------------------------------------
__global__ __launch_bounds__(64)
void k_head(const bf16_t* __restrict__ Kbf, const float* __restrict__ kvec,
            const float* __restrict__ hF, const float* __restrict__ critic_w,
            const float* __restrict__ critic_b, const int* __restrict__ actions,
            int t, float* __restrict__ out, int writeTail)
{
    const int n = blockIdx.x;
    const int l = threadIdx.x;

    __shared__ float kv[2 * H_DIM];
    for (int i = l; i < 2 * H_DIM; i += 64) kv[i] = kvec[(long)n * (2 * H_DIM) + i];
    __syncthreads();

    const v8bf* krow8 = (const v8bf*)(Kbf + ((long)n * L_DIM + l) * (2 * H_DIM));
    float s = 0.0f;
#pragma unroll 4
    for (int c = 0; c < (2 * H_DIM) / 8; ++c) {
        v8bf kk = krow8[c];
#pragma unroll
        for (int u = 0; u < 8; ++u) s += (float)kk[u] * kv[c * 8 + u];
    }

    __shared__ float sw[64];
    sw[l] = s;
    __syncthreads();
    float mx = -1e30f;
    for (int i = 0; i < 64; ++i) mx = fmaxf(mx, sw[i]);
    float e = expf(s - mx);
    __shared__ float se[64];
    se[l] = e;
    __syncthreads();
    float den = 0.0f;
    for (int i = 0; i < 64; ++i) den += se[i];
    float prob = e / den;

    int a = actions[(long)t * N_DIM + n];
    float* o = out + ((long)t * N_DIM + n) * OUT_STRIDE;
    o[258 + l] = prob;
    o[322 + l] = (l == a) ? 1.0f : 0.0f;

    float v = 0.0f;
    if (l == 0) {
        v = critic_b[0];
        for (int j = 0; j < H_DIM; ++j) v += hF[(long)n * H_DIM + j] * critic_w[j];
        o[0] = (float)a;
        o[1] = v;
    }

    if (writeTail) {
        float* tl = out + ((long)T_DIM * N_DIM + n) * OUT_STRIDE;
        tl[258 + l] = prob;
        tl[322 + l] = (l == a) ? 1.0f : 0.0f;
        for (int j = l; j < H_DIM; j += 64) tl[2 + j] = hF[(long)n * H_DIM + j];
        if (l == 0) { tl[0] = (float)a; tl[1] = v; }
    }
}

// ---------------------------------------------------------------------------
extern "C" void kernel_launch(void* const* d_in, const int* in_sizes, int n_in,
                              void* d_out, int out_size, void* d_ws, size_t ws_size,
                              hipStream_t stream)
{
    (void)in_sizes; (void)n_in; (void)out_size; (void)ws_size;

    const float* condition = (const float*)d_in[0];
    const float* emb    = (const float*)d_in[1];
    const float* wih_f  = (const float*)d_in[2];
    const float* whh_f  = (const float*)d_in[3];
    const float* bih_f  = (const float*)d_in[4];
    const float* bhh_f  = (const float*)d_in[5];
    const float* wih_b  = (const float*)d_in[6];
    const float* whh_b  = (const float*)d_in[7];
    const float* bih_b  = (const float*)d_in[8];
    const float* bhh_b  = (const float*)d_in[9];
    const float* f_w0   = (const float*)d_in[10];
    const float* f_b0   = (const float*)d_in[11];
    const float* f_w1   = (const float*)d_in[12];
    const float* f_b1   = (const float*)d_in[13];
    const float* c_wih  = (const float*)d_in[14];
    const float* c_whh  = (const float*)d_in[15];
    const float* c_bih  = (const float*)d_in[16];
    const float* c_bhh  = (const float*)d_in[17];
    const float* actor_w  = (const float*)d_in[18];
    const float* actor_b  = (const float*)d_in[19];
    const float* critic_w = (const float*)d_in[20];
    const float* critic_b = (const float*)d_in[21];
    const int* lines    = (const int*)d_in[22];
    const int* actions  = (const int*)d_in[23];
    float* out = (float*)d_out;

    // ---- workspace carve ----
    char* p = (char*)d_ws;
    auto carve = [&](size_t bytes) -> void* {
        void* r = (void*)p;
        p += (bytes + 255) & ~(size_t)255;
        return r;
    };
    bf16_t* Mbf  = (bf16_t*)carve((size_t)N_DIM * L_DIM * H_DIM * 2);        // 33.5 MB
    bf16_t* Kbf  = (bf16_t*)carve((size_t)N_DIM * L_DIM * 2 * H_DIM * 2);    // 67 MB (L2 resident)
    bf16_t* Wihf = (bf16_t*)carve((size_t)3 * H_DIM * H_DIM * 2);
    bf16_t* Whhf = (bf16_t*)carve((size_t)3 * H_DIM * H_DIM * 2);
    bf16_t* Wihb = (bf16_t*)carve((size_t)3 * H_DIM * H_DIM * 2);
    bf16_t* Whhb = (bf16_t*)carve((size_t)3 * H_DIM * H_DIM * 2);
    bf16_t* Wcih = (bf16_t*)carve((size_t)3 * H_DIM * H_DIM * 2);
    bf16_t* Wchh = (bf16_t*)carve((size_t)3 * H_DIM * H_DIM * 2);
    bf16_t* Wf0  = (bf16_t*)carve((size_t)H_DIM * (C_DIM + H_DIM) * 2);
    bf16_t* Wf1  = (bf16_t*)carve((size_t)H_DIM * H_DIM * 2);
    bf16_t* Wact = (bf16_t*)carve((size_t)2 * H_DIM * H_DIM * 2);
    float*  hA   = (float*)carve((size_t)N_DIM * H_DIM * 4);   // BiGRU state (reused per dir)
    bf16_t* hAb  = (bf16_t*)carve((size_t)N_DIM * H_DIM * 2);
    float*  hc   = (float*)carve((size_t)N_DIM * H_DIM * 4);   // GRUCell state
    bf16_t* hcb  = (bf16_t*)carve((size_t)N_DIM * H_DIM * 2);
    float*  gi   = (float*)carve((size_t)N_DIM * 3 * H_DIM * 4);
    float*  gh   = (float*)carve((size_t)N_DIM * 3 * H_DIM * 4);
    bf16_t* x0   = (bf16_t*)carve((size_t)N_DIM * (C_DIM + H_DIM) * 2);
    bf16_t* x1   = (bf16_t*)carve((size_t)N_DIM * H_DIM * 2);
    bf16_t* x2   = (bf16_t*)carve((size_t)N_DIM * H_DIM * 2);
    float*  kvec = (float*)carve((size_t)N_DIM * 2 * H_DIM * 4);

    auto cvt = [&](const float* src, bf16_t* dst, int n) {
        k_f32_to_bf16<<<(n + 255) / 256, 256, 0, stream>>>(src, dst, n);
    };
    auto gemm = [&](const bf16_t* A, long lda, const bf16_t* W, const float* bias,
                    int M, int Nc, int K, float* oF, bf16_t* oB, int ldo, int relu) {
        dim3 g(M / 64, Nc / 64);
        k_wmma_gemm<<<g, dim3(256), 0, stream>>>(A, lda, W, bias, M, Nc, K, oF, oB, ldo, relu);
    };

    // ---- phase 0: convert weights to bf16 ----
    cvt(wih_f, Wihf, 3 * H_DIM * H_DIM);
    cvt(whh_f, Whhf, 3 * H_DIM * H_DIM);
    cvt(wih_b, Wihb, 3 * H_DIM * H_DIM);
    cvt(whh_b, Whhb, 3 * H_DIM * H_DIM);
    cvt(c_wih, Wcih, 3 * H_DIM * H_DIM);
    cvt(c_whh, Wchh, 3 * H_DIM * H_DIM);
    cvt(f_w0,  Wf0,  H_DIM * (C_DIM + H_DIM));
    cvt(f_w1,  Wf1,  H_DIM * H_DIM);
    cvt(actor_w, Wact, 2 * H_DIM * H_DIM);

    // ---- phase 1: embedding gather ----
    k_embed<<<(N_DIM * L_DIM * H_DIM) / 256, 256, 0, stream>>>(emb, lines, Mbf);

    const long ldM = (long)L_DIM * H_DIM;   // A row stride inside M for seq-major access

    // ---- phase 2: forward GRU scan ----
    hipMemsetAsync(hA, 0, (size_t)N_DIM * H_DIM * 4, stream);
    hipMemsetAsync(hAb, 0, (size_t)N_DIM * H_DIM * 2, stream);
    for (int l = 0; l < L_DIM; ++l) {
        gemm(Mbf + (long)l * H_DIM, ldM, Wihf, bih_f, N_DIM, 3 * H_DIM, H_DIM, gi, nullptr, 3 * H_DIM, 0);
        gemm(hAb, H_DIM, Whhf, bhh_f, N_DIM, 3 * H_DIM, H_DIM, gh, nullptr, 3 * H_DIM, 0);
        k_gru_gate<<<(N_DIM * H_DIM) / 256, 256, 0, stream>>>(gi, gh, hA, hAb, Kbf /*dir 0*/, l, nullptr);
    }

    // ---- phase 3: backward GRU scan ----
    hipMemsetAsync(hA, 0, (size_t)N_DIM * H_DIM * 4, stream);
    hipMemsetAsync(hAb, 0, (size_t)N_DIM * H_DIM * 2, stream);
    for (int s = 0; s < L_DIM; ++s) {
        int l = L_DIM - 1 - s;
        gemm(Mbf + (long)l * H_DIM, ldM, Wihb, bih_b, N_DIM, 3 * H_DIM, H_DIM, gi, nullptr, 3 * H_DIM, 0);
        gemm(hAb, H_DIM, Whhb, bhh_b, N_DIM, 3 * H_DIM, H_DIM, gh, nullptr, 3 * H_DIM, 0);
        k_gru_gate<<<(N_DIM * H_DIM) / 256, 256, 0, stream>>>(gi, gh, hA, hAb, Kbf + H_DIM /*dir 1*/, l, nullptr);
    }

    // ---- phase 4: time loop ----
    hipMemsetAsync(hc, 0, (size_t)N_DIM * H_DIM * 4, stream);
    hipMemsetAsync(hcb, 0, (size_t)N_DIM * H_DIM * 2, stream);
    for (int t = 0; t < T_DIM; ++t) {
        k_build_x0<<<(N_DIM * (C_DIM + H_DIM)) / 256, 256, 0, stream>>>(
            condition + (long)t * N_DIM * C_DIM, actions, t, Mbf, x0);
        gemm(x0, C_DIM + H_DIM, Wf0, f_b0, N_DIM, H_DIM, C_DIM + H_DIM, nullptr, x1, H_DIM, 1);
        gemm(x1, H_DIM, Wf1, f_b1, N_DIM, H_DIM, H_DIM, nullptr, x2, H_DIM, 1);
        gemm(x2, H_DIM, Wcih, c_bih, N_DIM, 3 * H_DIM, H_DIM, gi, nullptr, 3 * H_DIM, 0);
        gemm(hcb, H_DIM, Wchh, c_bhh, N_DIM, 3 * H_DIM, H_DIM, gh, nullptr, 3 * H_DIM, 0);
        k_gru_gate<<<(N_DIM * H_DIM) / 256, 256, 0, stream>>>(
            gi, gh, hc, hcb, nullptr, 0, out + ((long)t * N_DIM) * OUT_STRIDE + 2);
        gemm(hcb, H_DIM, Wact, actor_b, N_DIM, 2 * H_DIM, H_DIM, kvec, nullptr, 2 * H_DIM, 0);
        k_head<<<N_DIM, 64, 0, stream>>>(Kbf, kvec, hc, critic_w, critic_b, actions,
                                         t, out, (t == T_DIM - 1) ? 1 : 0);
    }
}
